// CrossModalFusion_21620865368734
// MI455X (gfx1250) — matrix operand order repaired and checked by reading, compiled, not run on hardware
//
#include <hip/hip_runtime.h>

// Problem constants (match reference)
constexpr int kH  = 256;
constexpr int kNH = 4;
constexpr int kDH = 64;   // kH / kNH
constexpr int kB  = 256;
constexpr int kLA = 128;
constexpr int kLK = 256;

typedef __bf16 bf16_t;
typedef __attribute__((ext_vector_type(16))) __bf16 v16bf;
typedef __attribute__((ext_vector_type(8)))  float  v8f;
typedef unsigned int __attribute__((ext_vector_type(4))) v4u;
typedef int          __attribute__((ext_vector_type(4))) v4i;
typedef int          __attribute__((ext_vector_type(8))) v8i;

union Frag {
  v16bf     v;
  bf16_t    h[16];
  unsigned  u[8];
};

__device__ __forceinline__ v8f wmma_bf16(v16bf a, v16bf b, v8f c) {
  // (neg_a, A, neg_b, B, c_mod, C, reuse_a, reuse_b)
  return __builtin_amdgcn_wmma_f32_16x16x32_bf16(false, a, false, b, (short)0, c,
                                                 false, false);
}

// float -> bf16: use the native conversion; backend should select v_cvt*bf16*.
__device__ __forceinline__ bf16_t f2bf(float f) { return (bf16_t)f; }

// A-fragment (16x32 bf16, MxK): per-lane uint index (pair of K values) for vgpr i
__device__ __forceinline__ int a_uoff(int i, int half) {
  // K pair start = 2*i + 16*half; divided by 2
  return i + 8 * half;
}

// ---------------------------------------------------------------------------
// Tensor Data Mover: async 2-D bf16 tile load (rows x cols, row stride in
// elements) from global memory to LDS at byte offset lds_byte_off.
// D# built per CDNA5 ISA 8.3/8.4; issued by one wave, tracked by TENSORcnt.
// ---------------------------------------------------------------------------
__device__ __forceinline__ void tdm_load_2d_bf16(unsigned lds_byte_off,
                                                 const void* gptr,
                                                 unsigned rows, unsigned cols,
                                                 unsigned row_stride_elems) {
  unsigned long long ga = (unsigned long long)(__SIZE_TYPE__)gptr;
  v4u g0;
  g0.x = 0x1u;                                   // count=1, user mode
  g0.y = lds_byte_off;                           // lds_addr [63:32]
  g0.z = (unsigned)(ga & 0xffffffffu);           // global_addr [95:64]
  g0.w = (unsigned)((ga >> 32) & 0x01ffffffu)    // global_addr [120:96]
         | 0x80000000u;                          // type=2 ("image") [127:126]
  v8i g1;
  g1[0] = 0x00010000;                            // data_size=1 (2 bytes)
  g1[1] = (int)((cols & 0xffffu) << 16);         // tensor_dim0[15:0] @ [63:48]
  g1[2] = (int)((cols >> 16) | ((rows & 0xffffu) << 16)); // dim0 hi | dim1 lo
  g1[3] = (int)((rows >> 16) | (cols << 16));    // dim1 hi | tile_dim0
  g1[4] = (int)(rows & 0xffffu);                 // tile_dim1 (tile_dim2 = 0)
  g1[5] = (int)row_stride_elems;                 // tensor_dim0_stride[31:0]
  g1[6] = 0;                                     // stride0 hi | stride1 lo
  g1[7] = 0;                                     // stride1 hi
  v4i gz = {0, 0, 0, 0};
#if defined(__clang_major__) && (__clang_major__ >= 23)
  v8i gz8 = {0, 0, 0, 0, 0, 0, 0, 0};
  __builtin_amdgcn_tensor_load_to_lds(g0, g1, gz, gz, gz8, 0);
#else
  __builtin_amdgcn_tensor_load_to_lds(g0, g1, gz, gz, 0);
#endif
}

// ---------------------------------------------------------------------------
// Kernel 0: zero the ctx accumulation buffer
// ---------------------------------------------------------------------------
__global__ void cmf_zero(float* __restrict__ p, int n) {
  int i = blockIdx.x * blockDim.x + threadIdx.x;
  if (i < n) p[i] = 0.0f;
}

// ---------------------------------------------------------------------------
// Kernel 1: QKV projection GEMM, bf16 WMMA, bias fused, bf16 output.
//   p = blockIdx.z selects: 0 -> Q = atom @ Wq^T, 1 -> K = kg @ Wk^T, 2 -> V.
//   Block: 256 threads (8 waves). Workgroup tile 128(M) x 64(N); wave tile 16x64.
// ---------------------------------------------------------------------------
__global__ void cmf_proj(const float* __restrict__ atom,
                         const float* __restrict__ kg,
                         const float* __restrict__ w_in,
                         const float* __restrict__ b_in,
                         bf16_t* __restrict__ Qb,
                         bf16_t* __restrict__ Kb,
                         bf16_t* __restrict__ Vb) {
  const int p = blockIdx.z;
  const int mblocks = (p == 0) ? (kB * kLA / 128) : (kB * kLK / 128);
  if ((int)blockIdx.y >= mblocks) return;

  const float* X   = (p == 0) ? atom : kg;
  bf16_t*      out = (p == 0) ? Qb : (p == 1 ? Kb : Vb);
  const int n0 = blockIdx.x * 64;
  const int wrow0 = p * kH + n0;

  // Stage 64 x 256 weight tile as bf16 (32 KB LDS), converting f32 -> bf16.
  __shared__ bf16_t sW[64 * kH];
  for (int idx = threadIdx.x; idx < 64 * kH; idx += blockDim.x) {
    int j = idx >> 8, k = idx & (kH - 1);
    sW[idx] = f2bf(w_in[(wrow0 + j) * kH + k]);
  }
  __syncthreads();
  const unsigned* sW_u = (const unsigned*)sW;

  const int wave = threadIdx.x >> 5;
  const int lane = threadIdx.x & 31;
  const int half = lane >> 4;
  const int mr   = lane & 15;
  const int m0   = blockIdx.y * 128 + wave * 16;
  const size_t xrow = (size_t)(m0 + mr) * kH;

  v8f zero = {};
  v8f acc[4];
  #pragma unroll
  for (int t = 0; t < 4; ++t) acc[t] = zero;

  #pragma unroll
  for (int kk = 0; kk < kH / 32; ++kk) {
    const int kbase = kk * 32;
    if (kk + 1 < kH / 32)
      __builtin_prefetch(X + xrow + kbase + 32, 0, 0);  // global_prefetch_b8

    Frag af;
    #pragma unroll
    for (int i = 0; i < 8; ++i) {
      int koff = 2 * i + 16 * half;
      float2 x2 = *reinterpret_cast<const float2*>(X + xrow + kbase + koff);
      af.h[2 * i]     = f2bf(x2.x);
      af.h[2 * i + 1] = f2bf(x2.y);
    }
    #pragma unroll
    for (int t = 0; t < 4; ++t) {
      Frag bfg;
      #pragma unroll
      for (int i = 0; i < 8; ++i)
        bfg.u[i] = sW_u[(t * 16 + mr) * (kH / 2) + kk * 16 + half * 8 + i];
      acc[t] = wmma_bf16(af.v, bfg.v, acc[t]);
    }
  }

  // Epilogue: D layout -> M = i + 8*half, N = mr (within 16x16 tile t)
  #pragma unroll
  for (int t = 0; t < 4; ++t) {
    const int n = n0 + t * 16 + mr;
    const float bias = b_in[p * kH + n];
    #pragma unroll
    for (int i = 0; i < 8; ++i) {
      const int m = m0 + i + 8 * half;
      out[(size_t)m * kH + n] = f2bf(acc[t][i] + bias);
    }
  }
}

// ---------------------------------------------------------------------------
// Kernel 2: attention per (head, batch). 256 threads = 8 waves; wave w owns
// query rows [16w, 16w+16). K/V tiles staged via Tensor Data Mover
// (tensor_load_to_lds + s_wait_tensorcnt), V then transposed LDS->LDS so
// ctx B-fragment K-pairs are contiguous. Scores via WMMA, half-wave shfl
// softmax, probs through LDS into A-layout, ctx via WMMA, column sums
// atomically accumulated into ctxsum[b*H + h*DH + d] (/LA deferred).
// Dynamic LDS: sK 32K + sV 32K + sVt 32K + sP 64K = 160 KB.
// ---------------------------------------------------------------------------
__global__ void cmf_attn(const bf16_t* __restrict__ Qb,
                         const bf16_t* __restrict__ Kb,
                         const bf16_t* __restrict__ Vb,
                         float* __restrict__ ctxsum) {
  const int h = blockIdx.x;
  const int b = blockIdx.y;

  extern __shared__ char smem[];
  bf16_t* sK  = (bf16_t*)smem;              // [LK][DH]   @ 0
  bf16_t* sV  = (bf16_t*)(smem + 32768);    // [LK][DH]   @ 32768 (TDM dest)
  bf16_t* sVt = (bf16_t*)(smem + 65536);    // [DH][LK]   @ 65536 (transposed)
  bf16_t* sP  = (bf16_t*)(smem + 98304);    // [LA][LK]   @ 98304

  // --- TDM: async-stage K and V tiles (256 x 64 bf16, row stride H) ---
  if (threadIdx.x < 32) {
    const bf16_t* ksrc = Kb + (size_t)(b * kLK) * kH + h * kDH;
    const bf16_t* vsrc = Vb + (size_t)(b * kLK) * kH + h * kDH;
    tdm_load_2d_bf16(0u,     ksrc, kLK, kDH, kH);
    tdm_load_2d_bf16(32768u, vsrc, kLK, kDH, kH);
    __builtin_amdgcn_s_wait_tensorcnt(0);
  }
  __syncthreads();

  // Transpose V in LDS so B-fragment pairs (consecutive K=key) are contiguous.
  for (int idx = threadIdx.x; idx < kLK * kDH; idx += blockDim.x) {
    int key = idx >> 6, d = idx & 63;
    sVt[d * kLK + key] = sV[idx];
  }
  __syncthreads();

  const int wave = threadIdx.x >> 5;
  const int lane = threadIdx.x & 31;
  const int half = lane >> 4;
  const int mr   = lane & 15;
  const int m0   = wave * 16;

  const unsigned* Qu   = (const unsigned*)Qb;
  const unsigned* sKu  = (const unsigned*)sK;
  const unsigned* sVtu = (const unsigned*)sVt;

  // ---- scores: S = Q_bh (16x64 per wave) x K_bh^T -> 16 x 256 ----
  v8f zero = {};
  v8f s[16];
  #pragma unroll
  for (int t = 0; t < 16; ++t) s[t] = zero;

  #pragma unroll
  for (int kk = 0; kk < kDH / 32; ++kk) {
    Frag af;
    #pragma unroll
    for (int i = 0; i < 8; ++i)
      af.u[i] = Qu[(size_t)(b * kLA + m0 + mr) * (kH / 2) + h * (kDH / 2) +
                   kk * 16 + a_uoff(i, half)];
    #pragma unroll
    for (int t = 0; t < 16; ++t) {
      Frag bfg;
      #pragma unroll
      for (int i = 0; i < 8; ++i)
        bfg.u[i] = sKu[(t * 16 + mr) * (kDH / 2) + kk * 16 + half * 8 + i];
      s[t] = wmma_bf16(af.v, bfg.v, s[t]);
    }
  }

  // ---- softmax over 256 keys per query row; rows split across lane halves ----
  const float scale = 0.125f;  // 1/sqrt(64)
  #pragma unroll
  for (int i = 0; i < 8; ++i) {
    float mx = -3.0e38f;
    #pragma unroll
    for (int t = 0; t < 16; ++t) { s[t][i] *= scale; mx = fmaxf(mx, s[t][i]); }
    #pragma unroll
    for (int d = 1; d < 16; d <<= 1) mx = fmaxf(mx, __shfl_xor(mx, d, 32));
    float sum = 0.0f;
    #pragma unroll
    for (int t = 0; t < 16; ++t) {
      float e = __expf(s[t][i] - mx);
      s[t][i] = e;
      sum += e;
    }
    #pragma unroll
    for (int d = 1; d < 16; d <<= 1) sum += __shfl_xor(sum, d, 32);
    const float inv = 1.0f / sum;
    const int m = m0 + i + 8 * half;  // D-layout row
    #pragma unroll
    for (int t = 0; t < 16; ++t)
      sP[m * kLK + t * 16 + mr] = f2bf(s[t][i] * inv);
  }
  __syncthreads();
  const unsigned* sPu = (const unsigned*)sP;

  // ---- ctx = P (16x256) x V (256x64) ----
  v8f c[4];
  #pragma unroll
  for (int t = 0; t < 4; ++t) c[t] = zero;

  #pragma unroll
  for (int kk = 0; kk < kLK / 32; ++kk) {
    Frag af;
    #pragma unroll
    for (int i = 0; i < 8; ++i)
      af.u[i] = sPu[(m0 + mr) * (kLK / 2) + kk * 16 + a_uoff(i, half)];
    #pragma unroll
    for (int t = 0; t < 4; ++t) {
      Frag bfg;
      #pragma unroll
      for (int i = 0; i < 8; ++i)
        bfg.u[i] = sVtu[(t * 16 + mr) * (kLK / 2) + kk * 16 + half * 8 + i];
      c[t] = wmma_bf16(af.v, bfg.v, c[t]);
    }
  }

  // ---- reduce ctx over this wave's 16 query rows, accumulate globally ----
  #pragma unroll
  for (int t = 0; t < 4; ++t) {
    float cs = 0.0f;
    #pragma unroll
    for (int i = 0; i < 8; ++i) cs += c[t][i];
    cs += __shfl_xor(cs, 16, 32);  // combine the two row-halves
    if (lane < 16)
      atomicAdd(&ctxsum[b * kH + h * kDH + t * 16 + lane], cs);
  }
}

// ---------------------------------------------------------------------------
// Kernel 3: out[b,n] = mean_l atom[b,l,n] + (ctxsum[b,:]/LA) . out_w[n,:] + b_out[n]
// (out-projection commutes with the mean over LA — exact algebraic rewrite)
// ---------------------------------------------------------------------------
__global__ void cmf_final(const float* __restrict__ atom,
                          const float* __restrict__ w_out,
                          const float* __restrict__ b_out,
                          const float* __restrict__ ctxsum,
                          float* __restrict__ out) {
  const int b = blockIdx.x;
  const int n = threadIdx.x;

  __shared__ float sC[kH];
  sC[n] = ctxsum[b * kH + n];
  __syncthreads();

  float dot = 0.0f;
  #pragma unroll 8
  for (int k = 0; k < kH; ++k) dot += sC[k] * w_out[n * kH + k];

  float am = 0.0f;
  #pragma unroll 8
  for (int l = 0; l < kLA; ++l) am += atom[(size_t)(b * kLA + l) * kH + n];

  out[b * kH + n] = (am + dot) * (1.0f / (float)kLA) + b_out[n];
}

// ---------------------------------------------------------------------------
extern "C" void kernel_launch(void* const* d_in, const int* in_sizes, int n_in,
                              void* d_out, int out_size, void* d_ws, size_t ws_size,
                              hipStream_t stream) {
  (void)in_sizes; (void)n_in; (void)out_size; (void)ws_size;

  const float* atom  = (const float*)d_in[0];  // (B, LA, H)
  const float* kg    = (const float*)d_in[1];  // (B, LK, H)
  const float* w_in  = (const float*)d_in[2];  // (3H, H)
  const float* b_in  = (const float*)d_in[3];  // (3H,)
  const float* w_out = (const float*)d_in[4];  // (H, H)
  const float* b_out = (const float*)d_in[5];  // (H,)
  float* out = (float*)d_out;                  // (B, H)

  // Workspace layout (bytes):
  //   Qb: B*LA*H bf16 = 16,777,216
  //   Kb: B*LK*H bf16 = 33,554,432
  //   Vb: B*LK*H bf16 = 33,554,432
  //   ctxsum: B*H f32 =    262,144     (total ~84.1 MB)
  char* ws = (char*)d_ws;
  bf16_t* Qb = (bf16_t*)ws;
  bf16_t* Kb = (bf16_t*)(ws + 16777216ull);
  bf16_t* Vb = (bf16_t*)(ws + 16777216ull + 33554432ull);
  float* ctxsum = (float*)(ws + 16777216ull + 2ull * 33554432ull);

  cmf_zero<<<dim3((kB * kH + 255) / 256), 256, 0, stream>>>(ctxsum, kB * kH);

  // QKV projections: grid (N-tiles=4, M-tiles=512 max, proj=3)
  cmf_proj<<<dim3(kH / 64, (kB * kLK) / 128, 3), 256, 0, stream>>>(
      atom, kg, w_in, b_in, Qb, Kb, Vb);

  // Attention: one block per (head, batch); 160 KB dynamic LDS
  cmf_attn<<<dim3(kNH, kB), 256, 163840, stream>>>(Qb, Kb, Vb, ctxsum);

  // Final fused out-projection-of-mean + residual mean + bias
  cmf_final<<<dim3(kB), kH, 0, stream>>>(atom, w_out, b_out, ctxsum, out);
}
